// MoELayer_18674517803160
// MI455X (gfx1250) — matrix operand order repaired and checked by reading, not compiled, not run on hardware
//
#include <hip/hip_runtime.h>
#include <hip/hip_bf16.h>

// Problem constants (match reference)
constexpr int Bc = 2, Sc = 512, Hc = 1024, Ic = 4096, Ec = 8, Kc = 2;
constexpr int Tc = Bc * Sc;        // 1024 tokens
constexpr int CAP = Tc;            // worst-case tokens per expert
constexpr float LN_EPS = 1e-5f;

// Workspace layout (bytes)
constexpr size_t OFF_USAGE  = 0;                    // E floats
constexpr size_t OFF_COUNTS = 256;                  // E u32
constexpr size_t OFF_LISTS  = 512;                  // E*CAP u32 (t | k<<31)  = 32KB
constexpr size_t OFF_WGTS   = 512 + 32768;          // E*CAP f32             = 32KB
constexpr size_t OFF_XLN    = 128 * 1024;           // T*H bf16 = 2MB
constexpr size_t OFF_ACT    = 4u * 1024 * 1024;     // T*K*I bf16 = 16MB

typedef __bf16  v16bf __attribute__((ext_vector_type(16)));
typedef float   v8f   __attribute__((ext_vector_type(8)));

union Frag16 { v16bf v; __bf16 h[16]; uint4 q[2]; };

__device__ __forceinline__ float gelu_exact(float x) {
    return 0.5f * x * (1.0f + erff(x * 0.70710678118654752f));
}

// --- CDNA5 async VMEM->LDS copy (ASYNCcnt-tracked, no VGPR round trip) -----
// Per-lane: copies 16 bytes from the lane's global address into the lane's
// LDS address. LDS byte offset = low 32 bits of the flat shared-memory addr.
__device__ __forceinline__ void async_copy16(void* lds_dst, const void* gsrc) {
    const unsigned lds = (unsigned)(uintptr_t)lds_dst;
    asm volatile("global_load_async_to_lds_b128 %0, %1, off"
                 :: "v"(lds), "v"(gsrc) : "memory");
}
__device__ __forceinline__ void wait_async0() {
    asm volatile("s_wait_asynccnt 0x0" ::: "memory");
}

// ---------------------------------------------------------------------------
// Kernel A: LayerNorm + residual copy + router logits + softmax/usage + top-2
// one block (256 threads = 8 waves) per token
// ---------------------------------------------------------------------------
__global__ __launch_bounds__(256) void moe_ln_router(
    const float* __restrict__ hs, const float* __restrict__ lnw,
    const float* __restrict__ lnb, const float* __restrict__ Wr,
    const float* __restrict__ br,
    float* __restrict__ out, float* __restrict__ usage,
    unsigned* __restrict__ counts, unsigned* __restrict__ lists,
    float* __restrict__ wgts, __bf16* __restrict__ xln)
{
    __shared__ float sx[Hc];      // normalized x (f32) for router
    __shared__ float sred[8];
    __shared__ float slog[Ec];

    const int t    = blockIdx.x;
    const int tid  = threadIdx.x;
    const int lane = tid & 31;
    const int wv   = tid >> 5;

    const float4 h4 = ((const float4*)(hs + (size_t)t * Hc))[tid];

    // mean
    float s = h4.x + h4.y + h4.z + h4.w;
    #pragma unroll
    for (int o = 16; o > 0; o >>= 1) s += __shfl_down(s, o, 32);
    if (lane == 0) sred[wv] = s;
    __syncthreads();
    float tot = 0.f;
    #pragma unroll
    for (int i = 0; i < 8; ++i) tot += sred[i];
    const float mean = tot * (1.0f / Hc);
    __syncthreads();

    // variance
    const float d0 = h4.x - mean, d1 = h4.y - mean, d2 = h4.z - mean, d3 = h4.w - mean;
    float s2 = d0 * d0 + d1 * d1 + d2 * d2 + d3 * d3;
    #pragma unroll
    for (int o = 16; o > 0; o >>= 1) s2 += __shfl_down(s2, o, 32);
    if (lane == 0) sred[wv] = s2;
    __syncthreads();
    tot = 0.f;
    #pragma unroll
    for (int i = 0; i < 8; ++i) tot += sred[i];
    const float inv = rsqrtf(tot * (1.0f / Hc) + LN_EPS);

    const float4 w4 = ((const float4*)lnw)[tid];
    const float4 b4 = ((const float4*)lnb)[tid];
    const float x0 = d0 * inv * w4.x + b4.x;
    const float x1 = d1 * inv * w4.y + b4.y;
    const float x2 = d2 * inv * w4.z + b4.z;
    const float x3 = d3 * inv * w4.w + b4.w;

    // residual copy to out
    ((float4*)(out + (size_t)t * Hc))[tid] = h4;

    // stash x for router + bf16 copy for WMMA A-operands
    sx[4 * tid + 0] = x0; sx[4 * tid + 1] = x1;
    sx[4 * tid + 2] = x2; sx[4 * tid + 3] = x3;
    union { __bf16 h[4]; uint2 u; } pk;
    pk.h[0] = (__bf16)x0; pk.h[1] = (__bf16)x1;
    pk.h[2] = (__bf16)x2; pk.h[3] = (__bf16)x3;
    *(uint2*)(xln + (size_t)t * Hc + 4 * tid) = pk.u;
    __syncthreads();

    // router: wave wv computes logit for expert wv (E == 8 waves)
    float acc = 0.f;
    const float* wr = Wr + (size_t)wv * Hc;
    for (int j = lane; j < Hc; j += 32) acc += sx[j] * wr[j];
    #pragma unroll
    for (int o = 16; o > 0; o >>= 1) acc += __shfl_down(acc, o, 32);
    if (lane == 0) slog[wv] = acc + br[wv];
    __syncthreads();

    if (tid == 0) {
        float lg[Ec], mx = -3.4e38f;
        #pragma unroll
        for (int e = 0; e < Ec; ++e) { lg[e] = slog[e]; mx = fmaxf(mx, lg[e]); }
        float den = 0.f, p[Ec];
        #pragma unroll
        for (int e = 0; e < Ec; ++e) { p[e] = __expf(lg[e] - mx); den += p[e]; }
        const float rden = 1.0f / den;
        #pragma unroll
        for (int e = 0; e < Ec; ++e) atomicAdd(&usage[e], p[e] * rden);
        // top-2 (distinct by construction)
        int i1 = 0;
        #pragma unroll
        for (int e = 1; e < Ec; ++e) if (lg[e] > lg[i1]) i1 = e;
        int i2 = (i1 == 0) ? 1 : 0;
        #pragma unroll
        for (int e = 0; e < Ec; ++e) if (e != i1 && lg[e] > lg[i2]) i2 = e;
        const float w1 = 1.0f / (1.0f + __expf(lg[i2] - lg[i1]));
        const float w2 = 1.0f - w1;
        unsigned p1 = atomicAdd(&counts[i1], 1u);
        lists[i1 * CAP + p1] = (unsigned)t;                 // k = 0
        wgts [i1 * CAP + p1] = w1;
        unsigned p2 = atomicAdd(&counts[i2], 1u);
        lists[i2 * CAP + p2] = (unsigned)t | 0x80000000u;   // k = 1
        wgts [i2 * CAP + p2] = w2;
    }
}

// ---------------------------------------------------------------------------
// Kernel B: gathered expert FFN stage 1: act = gelu(x @ W1[e]^T + b1[e])
// grid (I/128, CAP/16, E), block 256 = 8 waves; each wave owns a 16x16 tile.
// A tiles double-buffered in LDS via async copies; B streamed fp32->bf16.
// ---------------------------------------------------------------------------
__global__ __launch_bounds__(256) void moe_ffn1(
    const float* __restrict__ W1, const float* __restrict__ b1,
    const unsigned* __restrict__ counts, const unsigned* __restrict__ lists,
    const __bf16* __restrict__ xln, __bf16* __restrict__ act)
{
    const int e  = blockIdx.z;
    const int m0 = blockIdx.y * 16;
    const int cnt = (int)counts[e];
    if (m0 >= cnt) return;

    __shared__ __bf16 Atile[2][16][128];   // 8KB double buffer
    __shared__ unsigned stok[16], sslot[16];

    const int tid = threadIdx.x, lane = tid & 31, wv = tid >> 5;

    if (tid < 16) {
        int r = m0 + tid; if (r > cnt - 1) r = cnt - 1;
        const unsigned pkd = lists[e * CAP + r];
        const unsigned tk  = pkd & 0x7FFFFFFFu;
        stok[tid]  = tk;
        sslot[tid] = 2u * tk + (pkd >> 31);
    }
    __syncthreads();

    const int nBase = blockIdx.x * 128 + wv * 16;
    const int col   = nBase + (lane & 15);
    const int koff  = (lane >> 4) * 16;          // B operand: upper lanes take K+16
    const int ao    = (lane >> 4) * 16;          // A operand: upper lanes take K+16
    const int arow  = lane & 15;
    const float* wrow = W1 + ((size_t)e * Ic + col) * Hc;

    // per-thread staging slot: 16 rows x 128 cols bf16, 16B per thread
    const int srow = tid >> 4;
    const int scol = (tid & 15) * 8;
    const __bf16* arow_g = xln + (size_t)stok[srow] * Hc + scol;

    Frag16 afrag, bfrag;
    v8f acc = {};

    // prologue: stage chunk 0
    async_copy16(&Atile[0][srow][scol], arow_g);
    wait_async0();
    __syncthreads();

    int buf = 0;
    for (int kc = 0; kc < Hc; kc += 128) {
        if (kc + 128 < Hc) {
            async_copy16(&Atile[buf ^ 1][srow][scol], arow_g + kc + 128);
            __builtin_prefetch(wrow + kc + 128, 0, 1);
        }
        #pragma unroll
        for (int ks = 0; ks < 128; ks += 32) {
            afrag.q[0] = *(const uint4*)&Atile[buf][arow][ks + ao];
            afrag.q[1] = *(const uint4*)&Atile[buf][arow][ks + ao + 8];
            const float4* bp = (const float4*)(wrow + kc + ks + koff);
            #pragma unroll
            for (int q = 0; q < 4; ++q) {
                const float4 f = bp[q];
                bfrag.h[4 * q + 0] = (__bf16)f.x;
                bfrag.h[4 * q + 1] = (__bf16)f.y;
                bfrag.h[4 * q + 2] = (__bf16)f.z;
                bfrag.h[4 * q + 3] = (__bf16)f.w;
            }
            acc = __builtin_amdgcn_wmma_f32_16x16x32_bf16(
                false, afrag.v, false, bfrag.v, (short)0, acc, false, false);
        }
        wait_async0();
        __syncthreads();
        buf ^= 1;
    }

    const float bias = b1[(size_t)e * Ic + col];
    #pragma unroll
    for (int v = 0; v < 8; ++v) {
        const int r = v + ((lane >> 4) * 8);
        if (m0 + r < cnt) {
            const float g = gelu_exact(acc[v] + bias);
            act[(size_t)sslot[r] * Ic + col] = (__bf16)g;
        }
    }
}

// ---------------------------------------------------------------------------
// Kernel C: stage 2: out += w_k * (act @ W2[e]^T + b2[e])   (atomic scatter)
// grid (H/128, CAP/16, E), block 256 = 8 waves
// ---------------------------------------------------------------------------
__global__ __launch_bounds__(256) void moe_ffn2(
    const float* __restrict__ W2, const float* __restrict__ b2,
    const unsigned* __restrict__ counts, const unsigned* __restrict__ lists,
    const float* __restrict__ wgts, const __bf16* __restrict__ act,
    float* __restrict__ out)
{
    const int e  = blockIdx.z;
    const int m0 = blockIdx.y * 16;
    const int cnt = (int)counts[e];
    if (m0 >= cnt) return;

    __shared__ __bf16 Atile[2][16][128];
    __shared__ unsigned stok[16], sslot[16];
    __shared__ float swgt[16];

    const int tid = threadIdx.x, lane = tid & 31, wv = tid >> 5;

    if (tid < 16) {
        int r = m0 + tid; if (r > cnt - 1) r = cnt - 1;
        const unsigned pkd = lists[e * CAP + r];
        const unsigned tk  = pkd & 0x7FFFFFFFu;
        stok[tid]  = tk;
        sslot[tid] = 2u * tk + (pkd >> 31);
        swgt[tid]  = wgts[e * CAP + r];
    }
    __syncthreads();

    const int nBase = blockIdx.x * 128 + wv * 16;
    const int col   = nBase + (lane & 15);
    const int koff  = (lane >> 4) * 16;
    const int ao    = (lane >> 4) * 16;
    const int arow  = lane & 15;
    const float* wrow = W2 + ((size_t)e * Hc + col) * Ic;

    const int srow = tid >> 4;
    const int scol = (tid & 15) * 8;
    const __bf16* arow_g = act + (size_t)sslot[srow] * Ic + scol;

    Frag16 afrag, bfrag;
    v8f acc = {};

    async_copy16(&Atile[0][srow][scol], arow_g);
    wait_async0();
    __syncthreads();

    int buf = 0;
    for (int kc = 0; kc < Ic; kc += 128) {
        if (kc + 128 < Ic) {
            async_copy16(&Atile[buf ^ 1][srow][scol], arow_g + kc + 128);
            __builtin_prefetch(wrow + kc + 128, 0, 1);
        }
        #pragma unroll
        for (int ks = 0; ks < 128; ks += 32) {
            afrag.q[0] = *(const uint4*)&Atile[buf][arow][ks + ao];
            afrag.q[1] = *(const uint4*)&Atile[buf][arow][ks + ao + 8];
            const float4* bp = (const float4*)(wrow + kc + ks + koff);
            #pragma unroll
            for (int q = 0; q < 4; ++q) {
                const float4 f = bp[q];
                bfrag.h[4 * q + 0] = (__bf16)f.x;
                bfrag.h[4 * q + 1] = (__bf16)f.y;
                bfrag.h[4 * q + 2] = (__bf16)f.z;
                bfrag.h[4 * q + 3] = (__bf16)f.w;
            }
            acc = __builtin_amdgcn_wmma_f32_16x16x32_bf16(
                false, afrag.v, false, bfrag.v, (short)0, acc, false, false);
        }
        wait_async0();
        __syncthreads();
        buf ^= 1;
    }

    const float bias = b2[(size_t)e * Hc + col];
    #pragma unroll
    for (int v = 0; v < 8; ++v) {
        const int r = v + ((lane >> 4) * 8);
        if (m0 + r < cnt) {
            atomicAdd(&out[(size_t)stok[r] * Hc + col], swgt[r] * (acc[v] + bias));
        }
    }
}

// ---------------------------------------------------------------------------
// Kernel D: balance loss = E * sum(usage^2) - 1, usage = mean prob per expert
// ---------------------------------------------------------------------------
__global__ void moe_balance(const float* __restrict__ usage, float* __restrict__ loss)
{
    if (threadIdx.x == 0 && blockIdx.x == 0) {
        float s = 0.f;
        #pragma unroll
        for (int e = 0; e < Ec; ++e) {
            const float u = usage[e] * (1.0f / (float)Tc);
            s += u * u;
        }
        loss[0] = (float)Ec * s - 1.0f;
    }
}

extern "C" void kernel_launch(void* const* d_in, const int* in_sizes, int n_in,
                              void* d_out, int out_size, void* d_ws, size_t ws_size,
                              hipStream_t stream) {
    const float* hs  = (const float*)d_in[0];
    const float* lnw = (const float*)d_in[1];
    const float* lnb = (const float*)d_in[2];
    const float* Wr  = (const float*)d_in[3];
    const float* br  = (const float*)d_in[4];
    const float* W1  = (const float*)d_in[5];
    const float* b1  = (const float*)d_in[6];
    const float* W2  = (const float*)d_in[7];
    const float* b2  = (const float*)d_in[8];

    char* ws = (char*)d_ws;
    float*    usage  = (float*)(ws + OFF_USAGE);
    unsigned* counts = (unsigned*)(ws + OFF_COUNTS);
    unsigned* lists  = (unsigned*)(ws + OFF_LISTS);
    float*    wgts   = (float*)(ws + OFF_WGTS);
    __bf16*   xln    = (__bf16*)(ws + OFF_XLN);
    __bf16*   actb   = (__bf16*)(ws + OFF_ACT);
    float*    out    = (float*)d_out;

    // zero usage + counts (graph-capture legal)
    hipMemsetAsync(ws, 0, 512, stream);

    moe_ln_router<<<Tc, 256, 0, stream>>>(hs, lnw, lnb, Wr, br,
                                          out, usage, counts, lists, wgts, xln);

    dim3 g1(Ic / 128, CAP / 16, Ec);   // (32, 64, 8)
    moe_ffn1<<<g1, 256, 0, stream>>>(W1, b1, counts, lists, xln, actb);

    dim3 g2(Hc / 128, CAP / 16, Ec);   // (8, 64, 8)
    moe_ffn2<<<g2, 256, 0, stream>>>(W2, b2, counts, lists, wgts, actb, out);

    moe_balance<<<1, 32, 0, stream>>>(usage, out + (size_t)Tc * Hc);
}
